// TacitReasoner_46557445489436
// MI455X (gfx1250) — compile-verified, matrix operand order checked by reading, not run on hardware
//
#include <hip/hip_runtime.h>

typedef _Float16 h16;
typedef __attribute__((ext_vector_type(16))) _Float16 v16h;
typedef __attribute__((ext_vector_type(8)))  _Float16 v8h;
typedef __attribute__((ext_vector_type(8)))  float    v8f;
typedef __attribute__((ext_vector_type(4)))  float    v4f;

#define TOTW 448
#define EMB  128

// ---------- CDNA5 async global->LDS copy (ASYNCcnt-tracked) ----------
// ISA 15.18.3 op 98: GLOBAL_LOAD_ASYNC_TO_LDS_B128  vdst(LDS byte addr), vaddr(64b), off
__device__ __forceinline__ void async_g2l_b128(unsigned lds_off, const void* gsrc) {
    asm volatile("global_load_async_to_lds_b128 %0, %1, off"
                 :: "v"(lds_off), "v"((unsigned long long)gsrc)
                 : "memory");
}
__device__ __forceinline__ void wait_async0() {
    asm volatile("s_wait_asynccnt 0x0" ::: "memory");
}
template <typename T>
__device__ __forceinline__ unsigned lds_offset(const T* p) {
    typedef __attribute__((address_space(3))) const T AST;
    return (unsigned)(unsigned long long)(AST*)p;
}

// Load one 16x32 (A, lane=row) or 32x16 (B, lane=col) f16 WMMA fragment.
// ISA 7.12.2: lanes 0-15 hold K=[0..7]+[16..23], lanes 16-31 K=[8..15]+[24..31]
//   -> two 16B chunks at +8*hi and +8*hi+16.
__device__ __forceinline__ v16h ld_frag(const h16* rowptr, int hi) {
    const h16* p = rowptr + 8 * hi;
    v8h lo = *(const v8h*)(p);
    v8h hh = *(const v8h*)(p + 16);
    return __builtin_shufflevector(lo, hh, 0,1,2,3,4,5,6,7,8,9,10,11,12,13,14,15);
}

// ---------- fused fp32 -> f16 weight conversion (all 7 matrices) ----------
struct CvtJobs {
    const float* src[7];
    h16*         dst[7];
    int          off[8];   // prefix sums, off[7] = total
};
__global__ __launch_bounds__(256) void k_cvt_all(CvtJobs j) {
    int i = blockIdx.x * blockDim.x + threadIdx.x;
    if (i >= j.off[7]) return;
    int s = 0;
#pragma unroll
    for (int t = 1; t < 7; ++t) s += (i >= j.off[t]);
    int local = i - j.off[s];
    j.dst[s][local] = (h16)j.src[s][local];
}

// base[b, 0:128]=emb[u[b]], base[b,128:256]=emb[v[b]]  (fp32 -> f16)
__global__ __launch_bounds__(256) void k_gather(const int* __restrict__ u,
                                                const int* __restrict__ v,
                                                const float* __restrict__ emb,
                                                h16* __restrict__ baseH, int B) {
    int idx = blockIdx.x * blockDim.x + threadIdx.x;   // one float4 per thread
    int total = B * 64;                                // 256 cols / 4
    if (idx >= total) return;
    int b = idx >> 6;
    int col = (idx & 63) * 4;
    const float* src = (col < EMB) ? (emb + (size_t)u[b] * EMB + col)
                                   : (emb + (size_t)v[b] * EMB + (col - EMB));
    v4f x = *(const v4f*)src;
    h16* d = baseH + (size_t)b * 256 + col;
    d[0] = (h16)x.x; d[1] = (h16)x.y; d[2] = (h16)x.z; d[3] = (h16)x.w;
}

// One witness stage: h = relu([base|prevW] @ W1^T + b1) ; w = sigmoid(h @ W2^T + b2)
// 32 batch rows / block, 8 waves. Activation tile async-staged into LDS as
// one contiguous [32 x KT] buffer (base cols 0..255, prevW cols 256..KT-1).
template <int KW, int N2>
__global__ __launch_bounds__(256) void k_witness(
    const h16* __restrict__ baseH, const h16* __restrict__ prevWH,
    const h16* __restrict__ W1h, const float* __restrict__ b1,
    const h16* __restrict__ W2h, const float* __restrict__ b2,
    h16* __restrict__ outWH, float* __restrict__ witF, int colOff) {
    constexpr int KT = 256 + KW;
    __shared__ h16 sX[32][KT];
    __shared__ h16 sH[32][64];

    const int tid = threadIdx.x;
    const int lane = tid & 31;
    const int wave = tid >> 5;
    const int ln = lane & 15, hi = lane >> 4;
    const int rowBase = blockIdx.x * 32;
    const int mt = wave >> 2, nt = wave & 3;

    // ---- async stage activation tile into LDS ----
    const unsigned sXoff = lds_offset(&sX[0][0]);
    for (int i = tid; i < 32 * 32; i += 256) {            // base: 32 rows x 32 16B-chunks
        int r = i >> 5, c = i & 31;
        async_g2l_b128(sXoff + (unsigned)(r * KT + c * 8) * 2,
                       baseH + (size_t)(rowBase + r) * 256 + c * 8);
    }
    if constexpr (KW > 0) {                               // prevW: 32 rows x KW/8 chunks
        constexpr int CW = KW / 8;
        for (int i = tid; i < 32 * CW; i += 256) {
            int r = i / CW, c = i % CW;
            async_g2l_b128(sXoff + (unsigned)(r * KT + 256 + c * 8) * 2,
                           prevWH + (size_t)(rowBase + r) * KW + c * 8);
        }
    }
    wait_async0();
    __syncthreads();

    // ---- layer 1: wave -> one 16x16 tile of [32 x 64] ----
    v8f acc = {};
    const h16* aRow = &sX[mt * 16 + ln][0];
    const h16* bRow = W1h + (size_t)(nt * 16 + ln) * KT;
#pragma unroll
    for (int k0 = 0; k0 < KT; k0 += 32) {
        v16h a = ld_frag(aRow + k0, hi);
        v16h b = ld_frag(bRow + k0, hi);
        acc = __builtin_amdgcn_wmma_f32_16x16x32_f16(false, a, false, b,
                                                     (short)0, acc, false, false);
    }
    float bias = b1[nt * 16 + ln];
#pragma unroll
    for (int r = 0; r < 8; ++r) {
        float vv = acc[r] + bias;
        sH[mt * 16 + r + 8 * hi][nt * 16 + ln] = (h16)(vv > 0.f ? vv : 0.f);
    }
    __syncthreads();

    // ---- layer 2: tiles 2 x (N2/16), wave-strided ----
    constexpr int T = 2 * (N2 / 16);
    for (int t = wave; t < T; t += 8) {
        int mt2 = t & 1, nt2 = t >> 1;
        v8f c = {};
        const h16* aR2 = &sH[mt2 * 16 + ln][0];
        const h16* bR2 = W2h + (size_t)(nt2 * 16 + ln) * 64;
#pragma unroll
        for (int k0 = 0; k0 < 64; k0 += 32) {
            v16h a = ld_frag(aR2 + k0, hi);
            v16h b = ld_frag(bR2 + k0, hi);
            c = __builtin_amdgcn_wmma_f32_16x16x32_f16(false, a, false, b,
                                                       (short)0, c, false, false);
        }
        int n2 = nt2 * 16 + ln;
        float bias2 = b2[n2];
#pragma unroll
        for (int r = 0; r < 8; ++r) {
            float x = c[r] + bias2;
            float s = 1.f / (1.f + __expf(-x));
            int row = rowBase + mt2 * 16 + r + 8 * hi;
            witF[(size_t)row * TOTW + colOff + n2] = s;
            if (outWH) outWH[(size_t)row * N2 + n2] = (h16)s;
        }
    }
}

// Tropical min-plus + decoder MLP. 16 rows / block, 128 threads (4 waves).
__global__ __launch_bounds__(128) void k_trop_dec(
    const float* __restrict__ witF, const float* __restrict__ tropW,
    const h16* __restrict__ dW1h, const float* __restrict__ dB1,
    const float* __restrict__ dW2, const float* __restrict__ dB2,
    float* __restrict__ out) {
    __shared__ float sWit[16][TOTW];
    __shared__ h16   sCode[16][128];
    __shared__ float sHid[16][64];

    const int tid = threadIdx.x;
    const int rowBase = blockIdx.x * 16;

    // async stage the 16x448 f32 wit tile (1792 16B chunks)
    const unsigned sWoff = lds_offset(&sWit[0][0]);
    const float* src = witF + (size_t)rowBase * TOTW;
    for (int i = tid; i < 16 * TOTW / 4; i += 128)
        async_g2l_b128(sWoff + (unsigned)i * 16, src + i * 4);
    wait_async0();
    __syncthreads();

    // one trop_w row per lane, 16-row wit tile broadcast from LDS
    const float* tw = tropW + (size_t)tid * TOTW;
    __builtin_prefetch(tw, 0, 1);
    float acc[16];
#pragma unroll
    for (int r = 0; r < 16; ++r) acc[r] = 3.4e38f;
    for (int j = 0; j < TOTW; j += 4) {
        v4f t4 = *(const v4f*)(tw + j);
#pragma unroll
        for (int r = 0; r < 16; ++r) {
            v4f w4 = *(const v4f*)(&sWit[r][j]);
            float m0 = fminf(w4.x + t4.x, w4.y + t4.y);
            float m1 = fminf(w4.z + t4.z, w4.w + t4.w);
            acc[r] = fminf(acc[r], fminf(m0, m1));
        }
    }
#pragma unroll
    for (int r = 0; r < 16; ++r) sCode[r][tid] = (h16)acc[r];
    __syncthreads();

    // decoder layer1: [16 x 128] @ [128 x 64] via WMMA, one 16x16 tile per wave
    const int lane = tid & 31, wave = tid >> 5;
    const int ln = lane & 15, hi = lane >> 4;
    v8f c = {};
    const h16* aRow = &sCode[ln][0];
    const h16* bRow = dW1h + (size_t)(wave * 16 + ln) * 128;
#pragma unroll
    for (int k0 = 0; k0 < 128; k0 += 32) {
        v16h a = ld_frag(aRow + k0, hi);
        v16h b = ld_frag(bRow + k0, hi);
        c = __builtin_amdgcn_wmma_f32_16x16x32_f16(false, a, false, b,
                                                   (short)0, c, false, false);
    }
    int n = wave * 16 + ln;
    float bias = dB1[n];
#pragma unroll
    for (int r = 0; r < 8; ++r) {
        float x = c[r] + bias;
        sHid[r + 8 * hi][n] = x > 0.f ? x : 0.f;
    }
    __syncthreads();

    if (tid < 16) {
        float s = dB2[0];
#pragma unroll 8
        for (int k = 0; k < 64; ++k) s += sHid[tid][k] * dW2[k];
        out[rowBase + tid] = s;
    }
}

extern "C" void kernel_launch(void* const* d_in, const int* in_sizes, int n_in,
                              void* d_out, int out_size, void* d_ws, size_t ws_size,
                              hipStream_t stream) {
    const int*   u    = (const int*)d_in[0];
    const int*   v    = (const int*)d_in[1];
    const float* emb  = (const float*)d_in[2];
    const float* e1w1 = (const float*)d_in[3];  const float* e1b1 = (const float*)d_in[4];
    const float* e1w2 = (const float*)d_in[5];  const float* e1b2 = (const float*)d_in[6];
    const float* e2w1 = (const float*)d_in[7];  const float* e2b1 = (const float*)d_in[8];
    const float* e2w2 = (const float*)d_in[9];  const float* e2b2 = (const float*)d_in[10];
    const float* e3w1 = (const float*)d_in[11]; const float* e3b1 = (const float*)d_in[12];
    const float* e3w2 = (const float*)d_in[13]; const float* e3b2 = (const float*)d_in[14];
    const float* trop = (const float*)d_in[15];
    const float* dw1  = (const float*)d_in[16]; const float* db1  = (const float*)d_in[17];
    const float* dw2  = (const float*)d_in[18]; const float* db2  = (const float*)d_in[19];

    const int B = in_sizes[0];   // 8192
    float* out = (float*)d_out;

    char* ws = (char*)d_ws;
    size_t off = 0;
    auto alloc = [&](size_t bytes) -> void* {
        void* p = ws + off;
        off = (off + bytes + 255) & ~(size_t)255;
        return p;
    };
    h16*   baseH = (h16*)alloc((size_t)B * 256 * 2);
    h16*   w1H   = (h16*)alloc((size_t)B * 64 * 2);
    h16*   w2H   = (h16*)alloc((size_t)B * 128 * 2);
    float* witF  = (float*)alloc((size_t)B * TOTW * 4);
    h16* e1w1H = (h16*)alloc(64 * 256 * 2);
    h16* e1w2H = (h16*)alloc(64 * 64 * 2);
    h16* e2w1H = (h16*)alloc(64 * 320 * 2);
    h16* e2w2H = (h16*)alloc(128 * 64 * 2);
    h16* e3w1H = (h16*)alloc(64 * 384 * 2);
    h16* e3w2H = (h16*)alloc(256 * 64 * 2);
    h16* dw1H  = (h16*)alloc(64 * 128 * 2);

    CvtJobs j;
    const float* srcs[7] = {e1w1, e1w2, e2w1, e2w2, e3w1, e3w2, dw1};
    h16*         dsts[7] = {e1w1H, e1w2H, e2w1H, e2w2H, e3w1H, e3w2H, dw1H};
    const int    ns[7]   = {64*256, 64*64, 64*320, 128*64, 64*384, 256*64, 64*128};
    int acc = 0;
    for (int t = 0; t < 7; ++t) {
        j.src[t] = srcs[t]; j.dst[t] = dsts[t]; j.off[t] = acc; acc += ns[t];
    }
    j.off[7] = acc;
    k_cvt_all<<<(acc + 255) / 256, 256, 0, stream>>>(j);

    k_gather<<<(B * 64 + 255) / 256, 256, 0, stream>>>(u, v, emb, baseH, B);

    k_witness<0, 64><<<B / 32, 256, 0, stream>>>(
        baseH, nullptr, e1w1H, e1b1, e1w2H, e1b2, w1H, witF, 0);
    k_witness<64, 128><<<B / 32, 256, 0, stream>>>(
        baseH, w1H, e2w1H, e2b1, e2w2H, e2b2, w2H, witF, 64);
    k_witness<128, 256><<<B / 32, 256, 0, stream>>>(
        baseH, w2H, e3w1H, e3b1, e3w2H, e3b2, nullptr, witF, 192);

    k_trop_dec<<<B / 16, 128, 0, stream>>>(witF, trop, dw1H, db1, dw2, db2, out);
}
